// Static_82300163326800
// MI455X (gfx1250) — matrix operand order repaired and checked by reading, not compiled, not run on hardware
//
#include <hip/hip_runtime.h>
#include <cmath>

typedef __attribute__((ext_vector_type(16))) _Float16 v16h;
typedef __attribute__((ext_vector_type(8)))  _Float16 v8h;
typedef __attribute__((ext_vector_type(8)))  float    v8f;

#define PI_F 3.14159265358979323846f

// ---------------- workspace layout (bytes, 256-aligned) ----------------
// peak usage ~104 MB
static const size_t H1_OFF   = 0;                       // f16 [1024][26][26][32] conv1 out, NHWC, relu'd
static const size_t H2_OFF   = 44302336;                // f32 [1024][9216] pooled conv2 (relu via atomic max)
static const size_t FC1_OFF  = H2_OFF + 37748736;       // f32 [1024][128]
static const size_t ANG_OFF  = FC1_OFF + 524288;        // f32 [1024][10] angles
static const size_t W2P_OFF  = ANG_OFF + 40960;         // f16 [9][4][32][16] conv2 W in WMMA-B layout
static const size_t FC1B_OFF = W2P_OFF + 36864;         // f16 [288][8][32][16] fc1 W in WMMA-B layout
static const size_t H2H_OFF  = FC1B_OFF + 2359296;      // f16 [1024][9216] (h2 converted once)

// ---------------- conv1: direct conv + relu -> NHWC f16 ----------------
__global__ void __launch_bounds__(256) conv1_kernel(const float* __restrict__ x,
                                                    const float* __restrict__ w,
                                                    const float* __restrict__ bias,
                                                    _Float16* __restrict__ h1) {
  int idx = blockIdx.x * blockDim.x + threadIdx.x;
  if (idx >= 1024 * 26 * 26 * 32) return;
  int c = idx & 31;
  int r = idx >> 5;
  int xx = r % 26; r /= 26;
  int yy = r % 26;
  int b  = r / 26;
  const float* xp = x + b * 784;
  const float* wp = w + c * 9;
  float s = bias[c];
#pragma unroll
  for (int ky = 0; ky < 3; ky++)
#pragma unroll
    for (int kx = 0; kx < 3; kx++)
      s += xp[(yy + ky) * 28 + xx + kx] * wp[ky * 3 + kx];
  h1[idx] = (_Float16)fmaxf(s, 0.f);   // h1[((b*26+y)*26+x)*32 + c]
}

// ------------- pack conv2_w [64,32,3,3] -> WMMA B-fragments -------------
// B packing used here (contiguous K, same order as A): lanes 0-15 elem i = B[K=i][N=lane],
//                           lanes 16-31 elem i = B[K=16+i][N=lane-16]
__global__ void __launch_bounds__(256) pack_w2(const float* __restrict__ w2,
                                               _Float16* __restrict__ dst) {
  int idx = blockIdx.x * blockDim.x + threadIdx.x;
  if (idx >= 9 * 4 * 32 * 16) return;
  int i    = idx & 15;
  int lane = (idx >> 4) & 31;
  int nc   = (idx >> 9) & 3;
  int t    = idx >> 11;
  int ky = t / 3, kx = t % 3;
  int c = (lane < 16) ? i : (16 + i);
  int o = nc * 16 + (lane & 15);
  dst[idx] = (_Float16)w2[((o * 32 + c) * 3 + ky) * 3 + kx];
}

// ------------- pack fc1_w [128,9216] -> WMMA B-fragments ---------------
__global__ void __launch_bounds__(256) pack_fc1(const float* __restrict__ w,
                                                _Float16* __restrict__ dst) {
  int idx = blockIdx.x * blockDim.x + threadIdx.x;
  if (idx >= 288 * 8 * 32 * 16) return;
  int i    = idx & 15;
  int lane = (idx >> 4) & 31;
  int nc   = (idx >> 9) & 7;
  int ks   = idx >> 12;
  int k = ks * 32 + ((lane < 16) ? i : (16 + i));
  int n = nc * 16 + (lane & 15);
  dst[idx] = (_Float16)w[n * 9216 + k];
}

// -------- conv2 via WMMA (implicit im2col) + bias + relu + maxpool -----
// one wave per tile of 16 output pixels x 64 channels; 9 taps x 4 nchunks WMMAs
__global__ void __launch_bounds__(256) conv2_wmma(const _Float16* __restrict__ h1,
                                                  const _Float16* __restrict__ w2p,
                                                  const float* __restrict__ b2,
                                                  float* __restrict__ h2) {
  int lane = threadIdx.x & 31;
  int wave = threadIdx.x >> 5;
  int tile = blockIdx.x * 8 + wave;       // 1024*36 tiles total
  int b  = tile / 36;
  int p0 = (tile % 36) * 16;
  int m  = lane & 15;                      // A row M = lane&15 on both halves
  int p  = p0 + m;
  int y  = p / 24, xq = p % 24;
  int cb = (lane < 16) ? 0 : 16;           // A layout: low lanes K 0-15, high lanes 16-31

  v8f acc[4];
#pragma unroll
  for (int nc = 0; nc < 4; nc++)
#pragma unroll
    for (int r = 0; r < 8; r++) acc[nc][r] = 0.f;

#pragma unroll
  for (int t = 0; t < 9; t++) {
    int ky = t / 3, kx = t % 3;
    const _Float16* ap = h1 + ((size_t)((b * 26 + (y + ky)) * 26 + (xq + kx))) * 32 + cb;
    v8h alo = *(const v8h*)ap;
    v8h ahi = *(const v8h*)(ap + 8);
    v16h a;
#pragma unroll
    for (int i = 0; i < 8; i++) { a[i] = alo[i]; a[8 + i] = ahi[i]; }
#pragma unroll
    for (int nc = 0; nc < 4; nc++) {
      v16h bm = *(const v16h*)(w2p + ((size_t)((t * 4 + nc) * 32 + lane)) * 16);
      acc[nc] = __builtin_amdgcn_wmma_f32_16x16x32_f16(false, a, false, bm,
                                                       (short)0, acc[nc], false, false);
    }
  }

  // C layout: lanes 0-15 elem r = C[M=r][N=lane]; lanes 16-31 elem r = C[M=8+r][N=lane-16]
  // relu + 2x2 maxpool fused: h2 pre-zeroed, non-negative f32 order == uint order
  int mrow = (lane < 16) ? 0 : 8;
#pragma unroll
  for (int nc = 0; nc < 4; nc++) {
    int n = nc * 16 + (lane & 15);
    float bb = b2[n];
#pragma unroll
    for (int r = 0; r < 8; r++) {
      int pm = p0 + mrow + r;
      int py = pm / 24, px = pm % 24;
      float v = fmaxf(acc[nc][r] + bb, 0.f);
      unsigned* dst = (unsigned*)&h2[(size_t)b * 9216 + n * 144 + (py >> 1) * 12 + (px >> 1)];
      atomicMax(dst, __float_as_uint(v));
    }
  }
}

// -------- h2 f32 -> f16, each element converted exactly once ----------
__global__ void __launch_bounds__(256) h2_to_half(const float* __restrict__ src,
                                                  _Float16* __restrict__ dst) {
  int idx = blockIdx.x * blockDim.x + threadIdx.x;   // 1179648 threads x 8 elems
  if (idx >= (1024 * 9216) / 8) return;
  const float* sp = src + (size_t)idx * 8;
  float4 f0 = *(const float4*)sp;
  float4 f1 = *(const float4*)(sp + 4);
  v8h o;
  o[0] = (_Float16)f0.x; o[1] = (_Float16)f0.y; o[2] = (_Float16)f0.z; o[3] = (_Float16)f0.w;
  o[4] = (_Float16)f1.x; o[5] = (_Float16)f1.y; o[6] = (_Float16)f1.z; o[7] = (_Float16)f1.w;
  *(v8h*)(dst + (size_t)idx * 8) = o;
}

// --------------- fc1 via WMMA: [1024,9216]x[9216,128] ------------------
// wave computes 16 rows x 32 cols (two N-chunks): A fragment reused x2
__global__ void __launch_bounds__(256) fc1_wmma(const _Float16* __restrict__ h2h,
                                                const _Float16* __restrict__ w1b,
                                                const float* __restrict__ bias,
                                                float* __restrict__ out) {
  int lane = threadIdx.x & 31;
  int wave = threadIdx.x >> 5;
  int tile = blockIdx.x * 8 + wave;       // 256 tiles
  int mt = tile >> 2;                     // 0..63 batch tile
  int ng = tile & 3;                      // N-chunk pair: {2*ng, 2*ng+1}
  const _Float16* row = h2h + (size_t)(mt * 16 + (lane & 15)) * 9216;
  int koff = (lane < 16) ? 0 : 16;        // A layout: low lanes K 0-15, high lanes 16-31

  v8f acc0, acc1;
#pragma unroll
  for (int r = 0; r < 8; r++) { acc0[r] = 0.f; acc1[r] = 0.f; }

  for (int ks = 0; ks < 288; ks++) {
    const _Float16* ap = row + ks * 32 + koff;
    v8h alo = *(const v8h*)ap;
    v8h ahi = *(const v8h*)(ap + 8);
    v16h a;
#pragma unroll
    for (int i = 0; i < 8; i++) { a[i] = alo[i]; a[8 + i] = ahi[i]; }
    v16h b0 = *(const v16h*)(w1b + ((size_t)(ks * 8 + 2 * ng) * 32 + lane) * 16);
    v16h b1 = *(const v16h*)(w1b + ((size_t)(ks * 8 + 2 * ng + 1) * 32 + lane) * 16);
    acc0 = __builtin_amdgcn_wmma_f32_16x16x32_f16(false, a, false, b0,
                                                  (short)0, acc0, false, false);
    acc1 = __builtin_amdgcn_wmma_f32_16x16x32_f16(false, a, false, b1,
                                                  (short)0, acc1, false, false);
  }

  int mrow = (lane < 16) ? 0 : 8;
#pragma unroll
  for (int h = 0; h < 2; h++) {
    int n = (2 * ng + h) * 16 + (lane & 15);
    float bb = bias[n];
    const v8f& acc = h ? acc1 : acc0;
#pragma unroll
    for (int r = 0; r < 8; r++) {
      int bo = mt * 16 + mrow + r;
      out[(size_t)bo * 128 + n] = fmaxf(acc[r] + bb, 0.f);
    }
  }
}

// ------------------- fc2 + sigmoid * 2pi -> angles ---------------------
__global__ void __launch_bounds__(256) fc2_sigmoid(const float* __restrict__ h,
                                                   const float* __restrict__ w,
                                                   const float* __restrict__ bias,
                                                   float* __restrict__ ang) {
  int idx = blockIdx.x * blockDim.x + threadIdx.x;
  if (idx >= 1024 * 10) return;
  int b = idx / 10, o = idx % 10;
  const float* hr = h + b * 128;
  const float* wr = w + o * 128;
  float s = bias[o];
  for (int n = 0; n < 128; n++) s += hr[n] * wr[n];
  ang[idx] = 2.f * PI_F / (1.f + __expf(-s));
}

// ===================== quantum circuit (state in LDS) ==================
__device__ __forceinline__ float2 cmul(float2 a, float2 b) {
  return make_float2(a.x * b.x - a.y * b.y, a.x * b.y + a.y * b.x);
}
__device__ __forceinline__ float2 cadd(float2 a, float2 b) {
  return make_float2(a.x + b.x, a.y + b.y);
}
// insert a zero bit at mask position
__device__ __forceinline__ unsigned insb(unsigned v, unsigned m) {
  return ((v & ~(m - 1u)) << 1) | (v & (m - 1u));
}
#define WM(w) (1u << (9 - (w)))   // reference: wire 0 is the MSB of the 10-bit index

struct M2 { float2 a, b, c, d; };

__device__ __forceinline__ M2 rx_m(float t) {
  float sn, cs; __sincosf(0.5f * t, &sn, &cs);
  M2 m; m.a = make_float2(cs, 0); m.b = make_float2(0, -sn);
  m.c = make_float2(0, -sn); m.d = make_float2(cs, 0); return m;
}
__device__ __forceinline__ M2 ry_m(float t) {
  float sn, cs; __sincosf(0.5f * t, &sn, &cs);
  M2 m; m.a = make_float2(cs, 0); m.b = make_float2(-sn, 0);
  m.c = make_float2(sn, 0); m.d = make_float2(cs, 0); return m;
}
__device__ __forceinline__ M2 rz_m(float t) {
  float sn, cs; __sincosf(0.5f * t, &sn, &cs);
  M2 m; m.a = make_float2(cs, -sn); m.b = make_float2(0, 0);
  m.c = make_float2(0, 0); m.d = make_float2(cs, sn); return m;
}
__device__ __forceinline__ M2 rot_m(float phi, float th, float om) {
  float sn, cs;  __sincosf(0.5f * th, &sn, &cs);
  float sp, cp;  __sincosf(0.5f * (phi + om), &sp, &cp);
  float sm, cm;  __sincosf(0.5f * (phi - om), &sm, &cm);
  M2 m;
  m.a = make_float2(cs * cp, -cs * sp);
  m.b = make_float2(-sn * cm, -sn * sm);
  m.c = make_float2(sn * cm, -sn * sm);
  m.d = make_float2(cs * cp, cs * sp);
  return m;
}
__device__ __forceinline__ M2 u2_m(float phi, float lam) {
  const float r = 0.7071067811865476f;
  float sl, cl;   __sincosf(lam, &sl, &cl);
  float sp, cp;   __sincosf(phi, &sp, &cp);
  float spl, cpl; __sincosf(phi + lam, &spl, &cpl);
  M2 m;
  m.a = make_float2(r, 0);
  m.b = make_float2(-r * cl, -r * sl);
  m.c = make_float2(r * cp, r * sp);
  m.d = make_float2(r * cpl, r * spl);
  return m;
}
__device__ __forceinline__ M2 u3_m(float th, float phi, float lam) {
  float sn, cs;   __sincosf(0.5f * th, &sn, &cs);
  float sl, cl;   __sincosf(lam, &sl, &cl);
  float sp, cp;   __sincosf(phi, &sp, &cp);
  float spl, cpl; __sincosf(phi + lam, &spl, &cpl);
  M2 m;
  m.a = make_float2(cs, 0);
  m.b = make_float2(-sn * cl, -sn * sl);
  m.c = make_float2(sn * cp, sn * sp);
  m.d = make_float2(cs * cpl, cs * spl);
  return m;
}

__device__ __forceinline__ void apply1(float2* s, int tid, int w, M2 u) {
  unsigned mask = WM(w);
  for (unsigned p = (unsigned)tid; p < 512; p += 256) {
    unsigned i0 = insb(p, mask), i1 = i0 | mask;
    float2 a = s[i0], b = s[i1];
    s[i0] = cadd(cmul(u.a, a), cmul(u.b, b));
    s[i1] = cadd(cmul(u.c, a), cmul(u.d, b));
  }
  __syncthreads();
}
__device__ __forceinline__ void capply1(float2* s, int tid, int wc, int wt, M2 u) {
  unsigned mc = WM(wc), mt = WM(wt);
  unsigned lo = mc < mt ? mc : mt, hi = mc < mt ? mt : mc;
  unsigned i0 = (insb(insb((unsigned)tid, lo), hi)) | mc;   // control bit = 1, target = 0
  unsigned i1 = i0 | mt;
  float2 a = s[i0], b = s[i1];
  s[i0] = cadd(cmul(u.a, a), cmul(u.b, b));
  s[i1] = cadd(cmul(u.c, a), cmul(u.d, b));
  __syncthreads();
}
__device__ __forceinline__ void phase1(float2* s, int tid, int w, float2 p0, float2 p1) {
  unsigned mask = WM(w);
  for (unsigned i = (unsigned)tid; i < 1024; i += 256)
    s[i] = cmul(s[i], (i & mask) ? p1 : p0);
  __syncthreads();
}
__device__ __forceinline__ void swap2(float2* s, int tid, int wa, int wb) {
  unsigned ma = WM(wa), mb = WM(wb);
  unsigned lo = ma < mb ? ma : mb, hi = ma < mb ? mb : ma;
  unsigned i = insb(insb((unsigned)tid, lo), hi);
  float2 t = s[i | ma]; s[i | ma] = s[i | mb]; s[i | mb] = t;
  __syncthreads();
}
__device__ __forceinline__ void sort3(unsigned& a, unsigned& b, unsigned& c) {
  unsigned t;
  if (a > b) { t = a; a = b; b = t; }
  if (b > c) { t = b; b = c; c = t; }
  if (a > b) { t = a; a = b; b = t; }
}
__device__ __forceinline__ void cswap3(float2* s, int tid, int wc, int wa, int wb) {
  unsigned mc = WM(wc), ma = WM(wa), mb = WM(wb);
  unsigned m0 = mc, m1 = ma, m2 = mb; sort3(m0, m1, m2);
  if (tid < 128) {
    unsigned i = insb(insb(insb((unsigned)tid, m0), m1), m2) | mc;
    float2 t = s[i | ma]; s[i | ma] = s[i | mb]; s[i | mb] = t;
  }
  __syncthreads();
}
__device__ __forceinline__ void ccx(float2* s, int tid, int wc1, int wc2, int wt) {
  unsigned mc1 = WM(wc1), mc2 = WM(wc2), mt = WM(wt);
  unsigned m0 = mc1, m1 = mc2, m2 = mt; sort3(m0, m1, m2);
  if (tid < 128) {
    unsigned i = insb(insb(insb((unsigned)tid, m0), m1), m2) | mc1 | mc2;
    float2 t = s[i]; s[i] = s[i | mt]; s[i | mt] = t;
  }
  __syncthreads();
}
__device__ __forceinline__ void multirz(float2* s, int tid, unsigned wmask, float theta) {
  float sn, cs; __sincosf(0.5f * theta, &sn, &cs);
  float2 pm = make_float2(cs, -sn), pp = make_float2(cs, sn);
  for (unsigned i = (unsigned)tid; i < 1024; i += 256)
    s[i] = cmul(s[i], (__popc(i & wmask) & 1) ? pp : pm);
  __syncthreads();
}

__global__ void __launch_bounds__(256) circuit_kernel(
    const float* __restrict__ angles, float* __restrict__ out,
    const float* q_rx, const float* q_rz3, const float* q_ps6, const float* q_rot7,
    const float* q_mrz8, const float* q_crx9, const float* q_cry10, const float* q_crz11,
    const float* q_u2, const float* q_u3) {
  __shared__ float2 s[1024];
  __shared__ float  red[256];
  __shared__ float  ang[10];
  __shared__ float  evs[10];
  int tid = threadIdx.x;
  int b = blockIdx.x;
  if (tid < 10) ang[tid] = angles[b * 10 + tid];
  for (int i = tid; i < 1024; i += 256) s[i] = make_float2(i == 0 ? 1.f : 0.f, 0.f);
  __syncthreads();

  const float2 Z0 = make_float2(0, 0), O1 = make_float2(1, 0);
  M2 X; X.a = Z0; X.b = O1; X.c = O1; X.d = Z0;
  M2 Ym; Ym.a = Z0; Ym.b = make_float2(0, -1); Ym.c = make_float2(0, 1); Ym.d = Z0;
  M2 Zm; Zm.a = O1; Zm.b = Z0; Zm.c = Z0; Zm.d = make_float2(-1, 0);

  for (int k = 0; k < 10; k++) apply1(s, tid, k, rx_m(q_rx[k]));
  for (int k = 0; k < 10; k++) {
    apply1(s, tid, k, ry_m(ang[k]));
    apply1(s, tid, 0, rx_m(ang[0]));
  }
  for (int k = 0; k < 9; k++) capply1(s, tid, k, k + 1, X);
  capply1(s, tid, 9, 0, X);
  apply1(s, tid, 1, rx_m(ang[1]));
  apply1(s, tid, 5, rx_m(-PI_F * 0.25f));
  apply1(s, tid, 2, ry_m(ang[2]));
  apply1(s, tid, 3, rz_m(ang[3]));
  phase1(s, tid, 4, O1, make_float2(0, 1));                                     // S
  phase1(s, tid, 5, O1, make_float2(0.7071067811865476f, 0.7071067811865476f)); // T
  apply1(s, tid, 6, rz_m(q_rz3[0]));
  { M2 SX; SX.a = make_float2(0.5f, 0.5f); SX.b = make_float2(0.5f, -0.5f);
    SX.c = make_float2(0.5f, -0.5f); SX.d = make_float2(0.5f, 0.5f);
    apply1(s, tid, 7, SX); }
  capply1(s, tid, 0, 5, Zm);   // CZ
  capply1(s, tid, 0, 5, X);    // CNOT
  capply1(s, tid, 0, 5, Ym);   // CY
  capply1(s, tid, 3, 8, Ym);   // CY
  swap2(s, tid, 2, 3);
  cswap3(s, tid, 4, 5, 6);
  ccx(s, tid, 8, 5, 0);        // Toffoli: controls 8,5 target 0
  { float sn, cs; __sincosf(q_ps6[0], &sn, &cs); phase1(s, tid, 8, O1, make_float2(cs, sn)); }
  { float sn, cs; __sincosf(ang[7], &sn, &cs);   phase1(s, tid, 7, O1, make_float2(cs, sn)); }
  apply1(s, tid, 4, rot_m(q_rot7[0], q_rot7[1], q_rot7[2]));
  apply1(s, tid, 5, rot_m(ang[6], ang[7], ang[8]));
  multirz(s, tid, WM(2) | WM(3) | WM(4) | WM(5) | WM(6), q_mrz8[0]);
  multirz(s, tid, WM(3) | WM(4) | WM(6) | WM(7) | WM(8), ang[5]);
  capply1(s, tid, 0, 1, rx_m(ang[6]));
  capply1(s, tid, 4, 5, rx_m(q_crx9[0]));
  capply1(s, tid, 0, 1, ry_m(ang[6]));
  capply1(s, tid, 4, 5, ry_m(q_cry10[0]));
  capply1(s, tid, 0, 1, rz_m(ang[6]));
  capply1(s, tid, 4, 5, rz_m(q_crz11[0]));
  capply1(s, tid, 5, 6, rot_m(-PI_F * 0.25f, PI_F * 0.25f, PI_F * 0.5f));
  capply1(s, tid, 7, 8, rot_m(ang[5], ang[6], ang[7]));
  { float sn, cs; __sincosf(PI_F / 7.f, &sn, &cs); phase1(s, tid, 1, O1, make_float2(cs, sn)); }
  { float sn, cs; __sincosf(ang[9], &sn, &cs);     phase1(s, tid, 2, O1, make_float2(cs, sn)); }
  apply1(s, tid, 8, u2_m(q_u2[0], q_u2[1]));
  apply1(s, tid, 4, u2_m(ang[0], ang[1]));
  apply1(s, tid, 5, u3_m(q_u3[0], q_u3[1], q_u3[2]));
  apply1(s, tid, 5, u3_m(ang[4], ang[5], ang[6]));
  capply1(s, tid, 1, 2, X);    // CX12

  // <Y_w> = sum over pairs 2*Im(conj(psi0)*psi1)
  for (int w = 0; w < 10; w++) {
    unsigned mask = WM(w);
    float part = 0.f;
    for (unsigned p = (unsigned)tid; p < 512; p += 256) {
      unsigned i0 = insb(p, mask), i1 = i0 | mask;
      float2 a = s[i0], bb = s[i1];
      part += 2.f * (a.x * bb.y - a.y * bb.x);
    }
    red[tid] = part; __syncthreads();
    for (int off = 128; off > 0; off >>= 1) {
      if (tid < off) red[tid] += red[tid + off];
      __syncthreads();
    }
    if (tid == 0) evs[w] = red[0];
    __syncthreads();
  }
  if (tid == 0) {
    float mx = evs[0];
    for (int o = 1; o < 10; o++) mx = fmaxf(mx, evs[o]);
    float se = 0.f;
    for (int o = 0; o < 10; o++) se += __expf(evs[o] - mx);
    float lse = __logf(se) + mx;
    for (int o = 0; o < 10; o++) out[b * 10 + o] = evs[o] - lse;
  }
}

// ========================= launcher ====================================
extern "C" void kernel_launch(void* const* d_in, const int* in_sizes, int n_in,
                              void* d_out, int out_size, void* d_ws, size_t ws_size,
                              hipStream_t stream) {
  const float* x      = (const float*)d_in[0];
  const float* w1     = (const float*)d_in[1];
  const float* b1     = (const float*)d_in[2];
  const float* w2     = (const float*)d_in[3];
  const float* b2     = (const float*)d_in[4];
  const float* fw1    = (const float*)d_in[5];
  const float* fb1    = (const float*)d_in[6];
  const float* fw2    = (const float*)d_in[7];
  const float* fb2    = (const float*)d_in[8];
  const float* q_rx   = (const float*)d_in[9];
  const float* q_rz3  = (const float*)d_in[10];
  const float* q_ps6  = (const float*)d_in[11];
  const float* q_rot7 = (const float*)d_in[12];
  const float* q_mrz8 = (const float*)d_in[13];
  const float* q_crx9 = (const float*)d_in[14];
  const float* q_cry10= (const float*)d_in[15];
  const float* q_crz11= (const float*)d_in[16];
  const float* q_u2   = (const float*)d_in[17];
  const float* q_u3   = (const float*)d_in[18];

  char* ws = (char*)d_ws;
  _Float16* h1   = (_Float16*)(ws + H1_OFF);
  float*    h2   = (float*)(ws + H2_OFF);
  float*    fc1o = (float*)(ws + FC1_OFF);
  float*    angs = (float*)(ws + ANG_OFF);
  _Float16* w2p  = (_Float16*)(ws + W2P_OFF);
  _Float16* fc1b = (_Float16*)(ws + FC1B_OFF);
  _Float16* h2h  = (_Float16*)(ws + H2H_OFF);

  // pooled conv2 buffer must be zero for fused relu/maxpool atomic-max
  hipMemsetAsync(h2, 0, (size_t)1024 * 9216 * sizeof(float), stream);

  conv1_kernel<<<(1024 * 26 * 26 * 32 + 255) / 256, 256, 0, stream>>>(x, w1, b1, h1);
  pack_w2<<<(9 * 4 * 32 * 16 + 255) / 256, 256, 0, stream>>>(w2, w2p);
  pack_fc1<<<(288 * 8 * 32 * 16 + 255) / 256, 256, 0, stream>>>(fw1, fc1b);
  conv2_wmma<<<4608, 256, 0, stream>>>(h1, w2p, b2, h2);    // 36864 wave-tiles
  h2_to_half<<<(1024 * 9216 / 8 + 255) / 256, 256, 0, stream>>>(h2, h2h);
  fc1_wmma<<<32, 256, 0, stream>>>(h2h, fc1b, fb1, fc1o);   // 256 wave-tiles, A reused x2
  fc2_sigmoid<<<40, 256, 0, stream>>>(fc1o, fw2, fb2, angs);
  circuit_kernel<<<1024, 256, 0, stream>>>(angs, (float*)d_out,
      q_rx, q_rz3, q_ps6, q_rot7, q_mrz8, q_crx9, q_cry10, q_crz11, q_u2, q_u3);
}
